// Head_257698038399
// MI455X (gfx1250) — matrix operand order repaired and checked by reading, not compiled, not run on hardware
//
#include <hip/hip_runtime.h>

// Problem constants (B,T,C,H) = (4,4096,1024,64)
#define BB 4
#define TT 4096
#define CC 1024
#define HH 64

typedef _Float16 f16;
typedef __attribute__((ext_vector_type(16))) _Float16 v16h;
typedef __attribute__((ext_vector_type(8)))  _Float16 v8h;
typedef __attribute__((ext_vector_type(8)))  float    v8f;
typedef __attribute__((ext_vector_type(4)))  unsigned int v4u;
typedef __attribute__((ext_vector_type(8)))  int v8i;
typedef __attribute__((ext_vector_type(4)))  int v4i;

#if defined(__has_builtin)
#if __has_builtin(__builtin_amdgcn_tensor_load_to_lds) && __has_builtin(__builtin_amdgcn_s_wait_tensorcnt)
#define HAVE_TDM 1
#endif
#endif
#ifndef HAVE_TDM
#define HAVE_TDM 0
#endif

union ALoad { v16h v; v8h p[2]; };

static __device__ __forceinline__ v8f wmma_f16(v16h a, v16h b, v8f c) {
  // D = A(16x32 f16) * B(32x16 f16) + C(16x16 f32)
  return __builtin_amdgcn_wmma_f32_16x16x32_f16(false, a, false, b, (short)0, c,
                                                false, false);
}

#if HAVE_TDM
// Build a 2D Tensor-DMA descriptor (D#) per CDNA5 ISA ch.8 and issue
// tensor_load_to_lds (6-arg clang-23 form). data_size = 2 bytes (f16).
static __device__ __forceinline__ void tdm_load_2d(const void* g, void* lds,
    unsigned td0, unsigned td1, unsigned long long st0,
    unsigned tile0, unsigned tile1)
{
  unsigned long long ga = (unsigned long long)(size_t)g;
  v4u g0;
  g0.x = 1u;                                    // [1:0] count=1, user desc
  g0.y = (unsigned)(size_t)lds;                 // [63:32] lds_addr (low 32 of LDS aperture ptr)
  g0.z = (unsigned)ga;                          // global_addr[31:0]
  g0.w = (unsigned)((ga >> 32) & 0x01FFFFFFu)   // global_addr[56:32]
         | 0x80000000u;                         // [127:126] type=2 ("image")
  v8i g1;
  g1[0] = (int)(1u << 16);                      // data_size=1 => 2 bytes; wg_mask=0
  g1[1] = (int)((td0 & 0xFFFFu) << 16);         // tensor_dim0[15:0] (bits 63:48)
  g1[2] = (int)(((td0 >> 16) & 0xFFFFu) | ((td1 & 0xFFFFu) << 16));
  g1[3] = (int)(((td1 >> 16) & 0xFFFFu) | ((tile0 & 0xFFFFu) << 16));
  g1[4] = (int)(tile1 & 0xFFFFu);               // tile_dim1; tile_dim2=0
  g1[5] = (int)(st0 & 0xFFFFFFFFull);           // tensor_dim0_stride[31:0]
  g1[6] = (int)((st0 >> 32) & 0xFFFFull);       // stride hi16; dim1_stride lo=0 (2D)
  g1[7] = 0;
  v4i z4 = {0, 0, 0, 0};                        // groups 2/3 unused (2D tensor)
  v8i z8 = {0, 0, 0, 0, 0, 0, 0, 0};            // trailing group (clang-23 form), unused
  __builtin_amdgcn_tensor_load_to_lds(g0, g1, z4, z4, z8, 0);
}
#endif

// ---------------------------------------------------------------------------
// Kernel 0: W[q,k,v] (C x H fp32, row-major) -> Wt (H x C f16) transposed.
// ---------------------------------------------------------------------------
__global__ void wt_convert(const float* __restrict__ Wq, const float* __restrict__ Wk,
                           const float* __restrict__ Wv,
                           f16* __restrict__ Wqt, f16* __restrict__ Wkt,
                           f16* __restrict__ Wvt)
{
  int idx = blockIdx.x * blockDim.x + threadIdx.x;
  if (idx >= HH * CC) return;
  int h = idx / CC, c = idx - h * CC;
  Wqt[idx] = (f16)Wq[c * HH + h];
  Wkt[idx] = (f16)Wk[c * HH + h];
  Wvt[idx] = (f16)Wv[c * HH + h];
}

// ---------------------------------------------------------------------------
// Kernel 1: fused QKV projection with WMMA.
// Grid (T/64, B), 128 threads (4 waves). Wave w computes a 16-row stripe.
// Outputs: Qh,Kh as [B][T][H] f16; V transposed as Vt [B][H][T] f16.
// ---------------------------------------------------------------------------
__global__ __launch_bounds__(128) void qkv_kernel(
    const float* __restrict__ x,
    const f16* __restrict__ Wqt, const f16* __restrict__ Wkt,
    const f16* __restrict__ Wvt,
    f16* __restrict__ Qh, f16* __restrict__ Kh, f16* __restrict__ Vt)
{
  const int wave = threadIdx.x >> 5, lane = threadIdx.x & 31;
  const int mrow = lane & 15, hl = lane >> 4;     // A-layout: M=lane%16, k half by lane/16
  const int qb = blockIdx.x, b = blockIdx.y;
  const int t0 = qb * 64 + wave * 16;

  const float* xrow = x + ((size_t)b * TT + t0 + mrow) * CC;

  const v8f vzero = {0.f, 0.f, 0.f, 0.f, 0.f, 0.f, 0.f, 0.f};
  v8f accQ[4], accK[4], accV[4];
#pragma unroll
  for (int t = 0; t < 4; ++t) { accQ[t] = vzero; accK[t] = vzero; accV[t] = vzero; }

  for (int kc = 0; kc < CC / 32; ++kc) {
    const int kbase = kc * 32;
    // A operand: x rows, fp32 -> f16. Lane half selects k-chunks {0-15} / {16-31}.
    const float4* xp0 = (const float4*)(xrow + kbase + hl * 16);
    const float4* xp1 = (const float4*)(xrow + kbase + hl * 16 + 8);
    float4 f0 = xp0[0], f1 = xp0[1], f2 = xp1[0], f3 = xp1[1];
    float c0[8] = {f0.x, f0.y, f0.z, f0.w, f1.x, f1.y, f1.z, f1.w};
    float c1[8] = {f2.x, f2.y, f2.z, f2.w, f3.x, f3.y, f3.z, f3.w};
    ALoad a;
#pragma unroll
    for (int i = 0; i < 8; ++i) { a.p[0][i] = (f16)c0[i]; a.p[1][i] = (f16)c1[i]; }

#pragma unroll
    for (int t = 0; t < 4; ++t) {
      // B operand: Wt[h][c], 16 contiguous k-values per lane.
      const size_t woff = (size_t)(t * 16 + mrow) * CC + kbase + hl * 16;
      v16h bq = *(const v16h*)(Wqt + woff);
      v16h bk = *(const v16h*)(Wkt + woff);
      v16h bv = *(const v16h*)(Wvt + woff);
      accQ[t] = wmma_f16(a.v, bq, accQ[t]);
      accK[t] = wmma_f16(a.v, bk, accK[t]);
      accV[t] = wmma_f16(a.v, bv, accV[t]);
    }
  }

  // D layout: row = r + 8*hl, col = t*16 + mrow.
#pragma unroll
  for (int t = 0; t < 4; ++t) {
    const int h = t * 16 + mrow;
#pragma unroll
    for (int r = 0; r < 8; ++r) {
      const int trow = t0 + r + 8 * hl;
      Qh[((size_t)b * TT + trow) * HH + h] = (f16)accQ[t][r];
      Kh[((size_t)b * TT + trow) * HH + h] = (f16)accK[t][r];
      Vt[((size_t)b * HH + h) * TT + trow] = (f16)accV[t][r];
    }
  }
}

// ---------------------------------------------------------------------------
// Kernel 2: fused flash-attention (causal), 64x64 tiles, online softmax.
// Grid (T/64, B), 128 threads. K/V staged to LDS via TDM, double-buffered.
// ---------------------------------------------------------------------------
static __device__ __forceinline__ void stage_kv(
    const f16* __restrict__ Kbase, const f16* __restrict__ Vbase, int j,
    f16 (*ldsK)[64], f16 (*ldsV)[64], int wave, int tid)
{
#if HAVE_TDM
  (void)tid;
  if (wave == 0) {
    // K tile: 64 rows (keys) x 64 f16, row stride HH.
    tdm_load_2d(Kbase + (size_t)j * 64 * HH, &ldsK[0][0],
                /*td0*/HH, /*td1*/TT, /*stride0*/HH, /*tile0*/HH, /*tile1*/64);
    // V tile from transposed V: 64 rows (h) x 64 f16 (keys), row stride TT.
    tdm_load_2d(Vbase + (size_t)j * 64, &ldsV[0][0],
                /*td0*/TT, /*td1*/HH, /*stride0*/TT, /*tile0*/64, /*tile1*/HH);
  }
#else
  const f16* gK = Kbase + (size_t)j * 64 * HH;
  const f16* gV = Vbase + (size_t)j * 64;
  for (int idx = tid; idx < 64 * HH / 8; idx += 128) {
    ((v8h*)&ldsK[0][0])[idx] = ((const v8h*)gK)[idx];
    int r = idx >> 3, c = idx & 7;
    *(v8h*)&ldsV[r][c * 8] = *(const v8h*)(gV + (size_t)r * TT + c * 8);
  }
  (void)wave;
#endif
}

static __device__ __forceinline__ void stage_wait(int wave)
{
#if HAVE_TDM
  if (wave == 0) __builtin_amdgcn_s_wait_tensorcnt(0);
#else
  (void)wave;
#endif
}

__global__ __launch_bounds__(128) void attn_kernel(
    const f16* __restrict__ Qh, const f16* __restrict__ Kh,
    const f16* __restrict__ Vt, float* __restrict__ out)
{
  __shared__ __attribute__((aligned(64))) f16 Kl[2][64][64];   // [buf][key][h]
  __shared__ __attribute__((aligned(64))) f16 Vl[2][64][64];   // [buf][h][key]
  __shared__ __attribute__((aligned(64))) f16 Pl[4][16][64];   // per-wave P staging

  const int wave = threadIdx.x >> 5, lane = threadIdx.x & 31;
  const int mrow = lane & 15, hl = lane >> 4;
  const int qb = blockIdx.x, b = blockIdx.y;

  // Q stripe for this wave, in WMMA A layout, loaded once.
  const f16* Qg = Qh + ((size_t)b * TT + qb * 64 + wave * 16 + mrow) * HH;
  ALoad qa[2];
#pragma unroll
  for (int ks = 0; ks < 2; ++ks) {
    qa[ks].p[0] = *(const v8h*)(Qg + ks * 32 + hl * 16);
    qa[ks].p[1] = *(const v8h*)(Qg + ks * 32 + hl * 16 + 8);
  }

  const v8f vzero = {0.f, 0.f, 0.f, 0.f, 0.f, 0.f, 0.f, 0.f};
  v8f o[4];
#pragma unroll
  for (int t = 0; t < 4; ++t) o[t] = vzero;
  float rmax[8], rsum[8];
#pragma unroll
  for (int r = 0; r < 8; ++r) { rmax[r] = -3.0e38f; rsum[r] = 0.f; }

  const f16* Kbase = Kh + (size_t)b * TT * HH;
  const f16* Vbase = Vt + (size_t)b * HH * TT;

  stage_kv(Kbase, Vbase, 0, Kl[0], Vl[0], wave, threadIdx.x);
  stage_wait(wave);
  __syncthreads();

  for (int j = 0; j <= qb; ++j) {
    const int cur = j & 1;
    if (j < qb)  // async prefetch of next K/V block into the other buffer
      stage_kv(Kbase, Vbase, j + 1, Kl[cur ^ 1], Vl[cur ^ 1], wave, threadIdx.x);

    // S = Q * K^T  (scaled), tiles over 64 key columns.
    v8f s[4];
#pragma unroll
    for (int t = 0; t < 4; ++t) {
      s[t] = vzero;
#pragma unroll
      for (int ks = 0; ks < 2; ++ks) {
        v16h kb = *(const v16h*)&Kl[cur][t * 16 + mrow][ks * 32 + hl * 16];
        s[t] = wmma_f16(qa[ks].v, kb, s[t]);
      }
    }

    const bool diag = (j == qb);
#pragma unroll
    for (int t = 0; t < 4; ++t)
#pragma unroll
      for (int r = 0; r < 8; ++r) {
        float val = s[t][r] * 0.125f;                       // 1/sqrt(H)
        if (diag && (t * 16 + mrow > wave * 16 + r + 8 * hl)) val = -3.0e38f;
        s[t][r] = val;
      }

    // Online softmax: per-row reduction across the 16 lanes of each half.
#pragma unroll
    for (int r = 0; r < 8; ++r) {
      float mx = fmaxf(fmaxf(s[0][r], s[1][r]), fmaxf(s[2][r], s[3][r]));
#pragma unroll
      for (int m = 1; m < 16; m <<= 1) mx = fmaxf(mx, __shfl_xor(mx, m, 32));
      const float mnew = fmaxf(rmax[r], mx);
      const float corr = __expf(rmax[r] - mnew);
      float lsum = 0.f;
#pragma unroll
      for (int t = 0; t < 4; ++t) { float p = __expf(s[t][r] - mnew); s[t][r] = p; lsum += p; }
#pragma unroll
      for (int m = 1; m < 16; m <<= 1) lsum += __shfl_xor(lsum, m, 32);
      rsum[r] = rsum[r] * corr + lsum;
      rmax[r] = mnew;
#pragma unroll
      for (int t = 0; t < 4; ++t) o[t][r] *= corr;
    }

    // P: D layout -> LDS (row-major) -> reload in A layout (same wave, DS in-order).
#pragma unroll
    for (int t = 0; t < 4; ++t)
#pragma unroll
      for (int r = 0; r < 8; ++r)
        Pl[wave][r + 8 * hl][t * 16 + mrow] = (f16)s[t][r];

    ALoad pa[2];
#pragma unroll
    for (int ks = 0; ks < 2; ++ks) {
      pa[ks].p[0] = *(const v8h*)&Pl[wave][mrow][ks * 32 + hl * 16];
      pa[ks].p[1] = *(const v8h*)&Pl[wave][mrow][ks * 32 + hl * 16 + 8];
    }

    // O += P * V   (V^T staged so B-operand reads are contiguous along keys)
#pragma unroll
    for (int t = 0; t < 4; ++t)
#pragma unroll
      for (int ks = 0; ks < 2; ++ks) {
        v16h vb = *(const v16h*)&Vl[cur][t * 16 + mrow][ks * 32 + hl * 16];
        o[t] = wmma_f16(pa[ks].v, vb, o[t]);
      }

    stage_wait(wave);     // prefetch complete before buffer flip
    __syncthreads();      // all waves done reading `cur` before it is reused
  }

  // Epilogue: normalize and store fp32 output.
#pragma unroll
  for (int r = 0; r < 8; ++r) rsum[r] = 1.0f / rsum[r];
#pragma unroll
  for (int t = 0; t < 4; ++t)
#pragma unroll
    for (int r = 0; r < 8; ++r) {
      const int row = qb * 64 + wave * 16 + r + 8 * hl;
      out[((size_t)b * TT + row) * HH + t * 16 + mrow] = o[t][r] * rsum[r];
    }
}

// ---------------------------------------------------------------------------
extern "C" void kernel_launch(void* const* d_in, const int* in_sizes, int n_in,
                              void* d_out, int out_size, void* d_ws, size_t ws_size,
                              hipStream_t stream)
{
  const float* x  = (const float*)d_in[0];
  const float* Wq = (const float*)d_in[1];
  const float* Wk = (const float*)d_in[2];
  const float* Wv = (const float*)d_in[3];
  float* out = (float*)d_out;

  f16* ws  = (f16*)d_ws;                    // ~6.7 MB of f16 scratch
  f16* Wqt = ws;
  f16* Wkt = Wqt + (size_t)HH * CC;
  f16* Wvt = Wkt + (size_t)HH * CC;
  f16* Qh  = Wvt + (size_t)HH * CC;
  f16* Kh  = Qh + (size_t)BB * TT * HH;
  f16* Vth = Kh + (size_t)BB * TT * HH;

  wt_convert<<<(HH * CC + 255) / 256, 256, 0, stream>>>(Wq, Wk, Wv, Wqt, Wkt, Wvt);
  qkv_kernel<<<dim3(TT / 64, BB), 128, 0, stream>>>(x, Wqt, Wkt, Wvt, Qh, Kh, Vth);
  attn_kernel<<<dim3(TT / 64, BB), 128, 0, stream>>>(Qh, Kh, Vth, out);

  (void)in_sizes; (void)n_in; (void)out_size; (void)ws_size;
}
